// MultiHeadLatentAttention_19318762897778
// MI455X (gfx1250) — matrix operand order repaired and checked by reading, not compiled, not run on hardware
//
#include <hip/hip_runtime.h>

// ---------------------------------------------------------------------------
// MLA (multi-head latent attention) forward for MI455X / gfx1250 (wave32).
// Heavy math: v_wmma_f32_16x16x32_bf16. GEMM B-panels staged through LDS by
// the Tensor Data Mover (tensor_load_to_lds + s_wait_tensorcnt), double
// buffered; A fragments double-buffered in registers.
// ---------------------------------------------------------------------------

typedef __bf16 bf16_t;
typedef bf16_t v16bf __attribute__((ext_vector_type(16)));
typedef float  v8f   __attribute__((ext_vector_type(8)));
typedef unsigned int u32x4 __attribute__((ext_vector_type(4)));
typedef int          i32x8 __attribute__((ext_vector_type(8)));
typedef int          i32x4 __attribute__((ext_vector_type(4)));

union FragB { v16bf v; uint4 q[2]; };

#if defined(__has_builtin)
#if __has_builtin(__builtin_amdgcn_tensor_load_to_lds) && \
    __has_builtin(__builtin_amdgcn_s_wait_tensorcnt)
#define HAVE_TDM 1
#endif
#endif

constexpr int Bc  = 2;
constexpr int Sc  = 2048;
constexpr int Dc  = 2048;
constexpr int Hc  = 16;
constexpr int DNc = 128;
constexpr int DRc = 64;
constexpr int DVc = 128;
constexpr int KVRc = 512;
constexpr int QRc  = 1536;
constexpr int BS   = Bc * Sc;          // 4096 token rows
constexpr int QW   = Hc * (DNc + DRc); // 3072
constexpr int KVW  = Hc * (DNc + DVc); // 4096
constexpr int DFULL = DNc + DRc;       // 192
constexpr float EPSc = 1.1920929e-07f;

__device__ __forceinline__ unsigned short f32_to_bf16(float f) {
  unsigned int u = __float_as_uint(f);
  unsigned int r = u + 0x7FFFu + ((u >> 16) & 1u);   // round to nearest even
  return (unsigned short)(r >> 16);
}

__device__ __forceinline__ v8f vzero8() {
  v8f v;
#pragma unroll
  for (int i = 0; i < 8; ++i) v[i] = 0.0f;
  return v;
}

// --------------------------- prep kernels ----------------------------------

__global__ __launch_bounds__(256) void convert_f32_to_bf16(
    const float* __restrict__ in, unsigned short* __restrict__ out, unsigned n) {
  unsigned i = blockIdx.x * 256u + threadIdx.x;
  if (i < n) out[i] = f32_to_bf16(in[i]);
}

// in: K x N (row major f32)  ->  out: N x K (row major bf16)
__global__ __launch_bounds__(256) void transpose_f32_to_bf16(
    const float* __restrict__ in, unsigned short* __restrict__ out, int K, int N) {
  __shared__ float tile[32][33];
  const int tx = threadIdx.x & 31;
  const int ty = threadIdx.x >> 5;          // 0..7
  const int bn = blockIdx.x * 32;
  const int bk = blockIdx.y * 32;
#pragma unroll
  for (int i = 0; i < 32; i += 8) {
    int k = bk + ty + i, n = bn + tx;
    if (k < K && n < N) tile[ty + i][tx] = in[(size_t)k * N + n];
  }
  __syncthreads();
#pragma unroll
  for (int i = 0; i < 32; i += 8) {
    int n = bn + ty + i, k = bk + tx;
    if (n < N && k < K) out[(size_t)n * K + k] = f32_to_bf16(tile[tx][ty + i]);
  }
}

// one block per row; rmsnorm over W columns (input row stride may be larger)
__global__ __launch_bounds__(256) void rmsnorm_to_bf16(
    const float* __restrict__ in, int inStride, const float* __restrict__ g,
    unsigned short* __restrict__ out, int W) {
  const int row = blockIdx.x;
  const float* r = in + (size_t)row * inStride;
  float acc = 0.0f;
  for (int i = threadIdx.x; i < W; i += 256) { float v = r[i]; acc += v * v; }
  __shared__ float sred[256];
  sred[threadIdx.x] = acc;
  __syncthreads();
  for (int o = 128; o > 0; o >>= 1) {
    if ((int)threadIdx.x < o) sred[threadIdx.x] += sred[threadIdx.x + o];
    __syncthreads();
  }
  const float rms = rsqrtf(sred[0] / (float)W + EPSc);
  unsigned short* orow = out + (size_t)row * W;
  for (int i = threadIdx.x; i < W; i += 256) orow[i] = f32_to_bf16(r[i] * rms * g[i]);
}

// --------------------------- TDM helper ------------------------------------
// Issue one Tensor-DMA load of a (rows x cols) bf16 tile (row stride
// 'strideElems' elements) from global into contiguous LDS (row major).
__device__ __forceinline__ void tdm_load_tile(unsigned short* ldsDst,
                                              const unsigned short* gsrc,
                                              int rows, int cols,
                                              int strideElems) {
#if defined(HAVE_TDM)
  const unsigned long long ga = (unsigned long long)(uintptr_t)gsrc;
  const unsigned lds = (unsigned)(uintptr_t)ldsDst;  // low 32b = LDS offset
  u32x4 g0;
  g0[0] = 1u;                                        // count=1 (valid user D#)
  g0[1] = lds;                                       // lds_addr (bytes)
  g0[2] = (unsigned)(ga & 0xFFFFFFFFu);              // global_addr[31:0]
  g0[3] = (unsigned)((ga >> 32) & 0x1FFFFFFu) | (2u << 30);  // [56:32] | type=2
  i32x8 g1;
  g1[0] = 1 << 16;                                   // data_size = 2 bytes
  g1[1] = (cols & 0xFFFF) << 16;                     // tensor_dim0[15:0]
  g1[2] = ((rows & 0xFFFF) << 16) | ((cols >> 16) & 0xFFFF);  // dim1 lo | dim0 hi
  g1[3] = ((cols & 0xFFFF) << 16) | ((rows >> 16) & 0xFFFF);  // tile_dim0 | dim1 hi
  g1[4] = rows & 0xFFFF;                             // tile_dim1 (tile_dim2=0)
  g1[5] = strideElems;                               // tensor_dim0_stride[31:0]
  g1[6] = 0;
  g1[7] = 0;
  i32x4 z4;
  z4[0] = 0; z4[1] = 0; z4[2] = 0; z4[3] = 0;
  i32x8 z8;
  z8[0] = 0; z8[1] = 0; z8[2] = 0; z8[3] = 0;
  z8[4] = 0; z8[5] = 0; z8[6] = 0; z8[7] = 0;
  // 6-arg form (clang-23 / therock-10.0 headers on this toolchain)
  __builtin_amdgcn_tensor_load_to_lds(g0, g1, z4, z4, z8, 0);
#else
  (void)ldsDst; (void)gsrc; (void)rows; (void)cols; (void)strideElems;
#endif
}

// --------------------------- WMMA GEMM -------------------------------------
// C[M x N] f32 = A[M x K] bf16 (row major) * Bt[N x K] bf16 (row major = B^T)
// block = 8 waves; wave computes a 32 x 64 strip (block tile 256 x 64).
// B panel (64 x 32 per k-step) staged in LDS by the TDM, double buffered.
// Requires M%256==0, N%64==0, K%32==0.
__global__ __launch_bounds__(256) void gemm_bf16_wmma(
    const unsigned short* __restrict__ A, const unsigned short* __restrict__ Bt,
    float* __restrict__ C, int M, int N, int K) {
  __shared__ __align__(128) unsigned short ldsB[2][64 * 32];  // 2 x 4KB
  const int lane = threadIdx.x & 31;
  const int wave = threadIdx.x >> 5;
  const int r0 = blockIdx.y * 256 + wave * 32;
  const int c0 = blockIdx.x * 64;
  const int nlo = lane & 15;
  const int hi  = lane >> 4;
  const int kb  = hi * 8;    // unused; A loads also use kb2
  const int kb2 = hi * 16;   // B element base within 32-chunk

  v8f acc0[4], acc1[4];
#pragma unroll
  for (int t = 0; t < 4; ++t) { acc0[t] = vzero8(); acc1[t] = vzero8(); }

  const unsigned short* Arow0 = A + (size_t)(r0 + nlo) * K;
  const unsigned short* Arow1 = A + (size_t)(r0 + 16 + nlo) * K;
  const unsigned short* Bpanel = Bt + (size_t)c0 * K;

#if defined(HAVE_TDM)
  if (wave == 0) tdm_load_tile(ldsB[0], Bpanel, 64, 32, K);
#else
  {  // cooperative copy fallback: 64 rows x 32 elems = 256 x b128
    const int row = threadIdx.x >> 2, q = threadIdx.x & 3;
    *(uint4*)(&ldsB[0][row * 32 + q * 8]) =
        *(const uint4*)(Bpanel + (size_t)row * K + q * 8);
  }
#endif
  FragB a0c, a1c;
  a0c.q[0] = *(const uint4*)(Arow0 + kb2);
  a0c.q[1] = *(const uint4*)(Arow0 + kb2 + 8);
  a1c.q[0] = *(const uint4*)(Arow1 + kb2);
  a1c.q[1] = *(const uint4*)(Arow1 + kb2 + 8);
#if defined(HAVE_TDM)
  if (wave == 0) __builtin_amdgcn_s_wait_tensorcnt(0);
#endif
  __syncthreads();

  int cur = 0;
  for (int k0 = 0; k0 < K; k0 += 32) {
    const bool has_next = (k0 + 32) < K;
    FragB a0n, a1n;
    if (has_next) {
#if defined(HAVE_TDM)
      if (wave == 0) tdm_load_tile(ldsB[cur ^ 1], Bpanel + k0 + 32, 64, 32, K);
#else
      const int row = threadIdx.x >> 2, q = threadIdx.x & 3;
      *(uint4*)(&ldsB[cur ^ 1][row * 32 + q * 8]) =
          *(const uint4*)(Bpanel + (size_t)row * K + k0 + 32 + q * 8);
#endif
      a0n.q[0] = *(const uint4*)(Arow0 + k0 + 32 + kb2);
      a0n.q[1] = *(const uint4*)(Arow0 + k0 + 40 + kb2);
      a1n.q[0] = *(const uint4*)(Arow1 + k0 + 32 + kb2);
      a1n.q[1] = *(const uint4*)(Arow1 + k0 + 40 + kb2);
    }
    const unsigned short* bb = ldsB[cur];
#pragma unroll
    for (int t = 0; t < 4; ++t) {
      FragB bf;
      bf.q[0] = *(const uint4*)(bb + (t * 16 + nlo) * 32 + kb2);
      bf.q[1] = *(const uint4*)(bb + (t * 16 + nlo) * 32 + kb2 + 8);
      acc0[t] = __builtin_amdgcn_wmma_f32_16x16x32_bf16(
          false, a0c.v, false, bf.v, (short)0, acc0[t], false, false);
      acc1[t] = __builtin_amdgcn_wmma_f32_16x16x32_bf16(
          false, a1c.v, false, bf.v, (short)0, acc1[t], false, false);
    }
    if (has_next) {
#if defined(HAVE_TDM)
      if (wave == 0) __builtin_amdgcn_s_wait_tensorcnt(0);
#endif
      __syncthreads();
      a0c = a0n; a1c = a1n;
      cur ^= 1;
    }
  }
#pragma unroll
  for (int t = 0; t < 4; ++t) {
    const int n = c0 + t * 16 + nlo;
#pragma unroll
    for (int v = 0; v < 8; ++v) {
      C[(size_t)(r0 + v + 8 * hi) * N + n]      = acc0[t][v];
      C[(size_t)(r0 + 16 + v + 8 * hi) * N + n] = acc1[t][v];
    }
  }
}

// --------------------------- RoPE / layout builders ------------------------

__device__ __forceinline__ float rope_val(const float* base, int i, int s) {
  const int ii  = (i < 32) ? i : i - 32;
  const float inv = __powf(10000.0f, -(float)ii / 32.0f);
  float sn, cs;
  __sincosf((float)s * inv, &sn, &cs);
  const float e0 = base[2 * ii], e1 = base[2 * ii + 1];
  return (i < 32) ? (e0 * cs - e1 * sn) : (e1 * cs + e0 * sn);
}

// q f32 (BS x 3072) -> q_full bf16 (B*H, S, 192) with RoPE on last 64 dims
__global__ __launch_bounds__(256) void build_qfull(
    const float* __restrict__ q, unsigned short* __restrict__ qf) {
  const unsigned idx = blockIdx.x * 256u + threadIdx.x;
  const unsigned total = (unsigned)Bc * Hc * Sc * DFULL;
  if (idx >= total) return;
  const int j = idx % DFULL;
  unsigned r = idx / DFULL;
  const int s = r % Sc;
  const unsigned bh = r / Sc;            // b*H + h
  const int h = bh % Hc;
  const int b = bh / Hc;
  const float* qrow = q + (size_t)(b * Sc + s) * QW + h * DFULL;
  float val = (j < DNc) ? qrow[j] : rope_val(qrow + DNc, j - DNc, s);
  qf[idx] = f32_to_bf16(val);
}

// kv f32 (BS x 4096), kvc f32 (BS x 576) -> k_full bf16 (B*H,S,192), Vt (B*H,128,S)
__global__ __launch_bounds__(256) void build_kfull_v(
    const float* __restrict__ kv, const float* __restrict__ kvc,
    unsigned short* __restrict__ kf, unsigned short* __restrict__ vt) {
  const unsigned idx = blockIdx.x * 256u + threadIdx.x;
  const unsigned total = (unsigned)Bc * Hc * Sc * (DFULL + DVc);  // 320/(bh,s)
  if (idx >= total) return;
  const int j = idx % (DFULL + DVc);
  unsigned r = idx / (DFULL + DVc);
  const int s = r % Sc;
  const unsigned bh = r / Sc;
  const int h = bh % Hc;
  const int b = bh / Hc;
  const size_t tok = (size_t)b * Sc + s;
  if (j < DFULL) {
    float val;
    if (j < DNc) {
      val = kv[tok * KVW + h * (DNc + DVc) + j];
    } else {
      const float* kr = kvc + tok * (KVRc + DRc) + KVRc;   // shared rope key
      val = rope_val(kr, j - DNc, s);
    }
    kf[((size_t)bh * Sc + s) * DFULL + j] = f32_to_bf16(val);
  } else {
    const int d = j - DFULL;
    vt[((size_t)bh * DVc + d) * Sc + s] =
        f32_to_bf16(kv[tok * KVW + h * (DNc + DVc) + DNc + d]);
  }
}

// --------------------------- attention -------------------------------------
// grid (S/128, H, B), block 256 (= 8 waves). Each wave owns 16 q rows and
// streams 32-key chunks with online softmax; P relayout C->A goes through LDS.
__global__ __launch_bounds__(256) void mla_attention(
    const unsigned short* __restrict__ qf,  // (B*H, S, 192)
    const unsigned short* __restrict__ kf,  // (B*H, S, 192)
    const unsigned short* __restrict__ vt,  // (B*H, 128, S)
    unsigned short* __restrict__ ao) {      // (B*S, H*128)
  __shared__ __align__(16) unsigned short plds[8][16 * 32];
  const int lane = threadIdx.x & 31;
  const int wave = threadIdx.x >> 5;
  const int h = blockIdx.y, b = blockIdx.z;
  const int bh = b * Hc + h;
  const int qw = blockIdx.x * 128 + wave * 16;   // first q row of this wave
  const int nlo = lane & 15;
  const int hi  = lane >> 4;
  const int kb  = hi * 8;
  const int kb2 = hi * 16;

  const unsigned short* Q  = qf + (size_t)bh * Sc * DFULL;
  const unsigned short* Km = kf + (size_t)bh * Sc * DFULL;
  const unsigned short* V  = vt + (size_t)bh * DVc * Sc;

  // preload the wave's Q tile: 16 rows x 192 = 6 A-fragments
  FragB aq[6];
  {
    const unsigned short* qrow = Q + (size_t)(qw + nlo) * DFULL;
#pragma unroll
    for (int c = 0; c < 6; ++c) {
      aq[c].q[0] = *(const uint4*)(qrow + c * 32 + kb2);
      aq[c].q[1] = *(const uint4*)(qrow + c * 32 + kb2 + 8);
    }
  }

  v8f Oacc[8];
#pragma unroll
  for (int t = 0; t < 8; ++t) Oacc[t] = vzero8();
  float mrow[8], lrow[8];
#pragma unroll
  for (int v = 0; v < 8; ++v) { mrow[v] = -3.0e38f; lrow[v] = 0.0f; }

  const float scale = 0.07216878364870322f;  // 1/sqrt(192)
  const int nchunks = (qw + 47) >> 5;        // keys 0 .. qw+15 inclusive

  for (int ck = 0; ck < nchunks; ++ck) {
    const int k0 = ck * 32;
    v8f s0 = vzero8(), s1 = vzero8();
#pragma unroll
    for (int c = 0; c < 6; ++c) {
      const unsigned short* kr0 = Km + (size_t)(k0 + nlo) * DFULL + c * 32 + kb2;
      const unsigned short* kr1 = Km + (size_t)(k0 + 16 + nlo) * DFULL + c * 32 + kb2;
      FragB b0, b1;
      b0.q[0] = *(const uint4*)(kr0); b0.q[1] = *(const uint4*)(kr0 + 8);
      b1.q[0] = *(const uint4*)(kr1); b1.q[1] = *(const uint4*)(kr1 + 8);
      s0 = __builtin_amdgcn_wmma_f32_16x16x32_bf16(false, aq[c].v, false, b0.v,
                                                   (short)0, s0, false, false);
      s1 = __builtin_amdgcn_wmma_f32_16x16x32_bf16(false, aq[c].v, false, b1.v,
                                                   (short)0, s1, false, false);
    }
    // prefetch the 8 V fragments for this chunk; loads overlap the softmax ALU
    FragB bv[8];
#pragma unroll
    for (int t = 0; t < 8; ++t) {
      const unsigned short* vr = V + (size_t)(t * 16 + nlo) * Sc + k0 + kb2;
      bv[t].q[0] = *(const uint4*)(vr);
      bv[t].q[1] = *(const uint4*)(vr + 8);
    }
    const bool need_mask = (k0 + 31) > qw;
#pragma unroll
    for (int v = 0; v < 8; ++v) {
      const int qrow_g = qw + v + 8 * hi;
      s0[v] *= scale; s1[v] *= scale;
      if (need_mask) {
        if (k0 + nlo      > qrow_g) s0[v] = -3.0e38f;
        if (k0 + 16 + nlo > qrow_g) s1[v] = -3.0e38f;
      }
    }
    float alpha[8];
#pragma unroll
    for (int v = 0; v < 8; ++v) {
      float mx = fmaxf(s0[v], s1[v]);
#pragma unroll
      for (int o = 1; o < 16; o <<= 1) mx = fmaxf(mx, __shfl_xor(mx, o, 32));
      const float mnew = fmaxf(mrow[v], mx);
      alpha[v] = __expf(mrow[v] - mnew);
      mrow[v] = mnew;
    }
#pragma unroll
    for (int v = 0; v < 8; ++v) {
      const float p0 = __expf(s0[v] - mrow[v]);
      const float p1 = __expf(s1[v] - mrow[v]);
      s0[v] = p0; s1[v] = p1;
      float rs = p0 + p1;
#pragma unroll
      for (int o = 1; o < 16; o <<= 1) rs += __shfl_xor(rs, o, 32);
      lrow[v] = lrow[v] * alpha[v] + rs;
    }
    // relayout P (C layout) -> A layout through this wave's LDS slab
    unsigned short* pw = plds[wave];
#pragma unroll
    for (int v = 0; v < 8; ++v) {
      const int m = v + 8 * hi;
      pw[m * 32 + nlo]      = f32_to_bf16(s0[v]);
      pw[m * 32 + 16 + nlo] = f32_to_bf16(s1[v]);
    }
    asm volatile("s_wait_dscnt 0x0" ::: "memory");
    FragB pa;
    pa.q[0] = *(const uint4*)(pw + nlo * 32 + kb2);
    pa.q[1] = *(const uint4*)(pw + nlo * 32 + kb2 + 8);
#pragma unroll
    for (int t = 0; t < 8; ++t) {
      v8f o = Oacc[t];
#pragma unroll
      for (int v = 0; v < 8; ++v) o[v] *= alpha[v];
      Oacc[t] = __builtin_amdgcn_wmma_f32_16x16x32_bf16(false, pa.v, false,
                                                        bv[t].v, (short)0, o,
                                                        false, false);
    }
  }
  // epilogue: normalize, write bf16 to (B*S, H*128)
#pragma unroll
  for (int t = 0; t < 8; ++t) {
#pragma unroll
    for (int v = 0; v < 8; ++v) {
      const int qr = qw + v + 8 * hi;
      const float val = Oacc[t][v] / lrow[v];
      ao[((size_t)b * Sc + qr) * (Hc * DVc) + h * DVc + t * 16 + nlo] =
          f32_to_bf16(val);
    }
  }
}

// --------------------------- host orchestration ----------------------------

extern "C" void kernel_launch(void* const* d_in, const int* in_sizes, int n_in,
                              void* d_out, int out_size, void* d_ws, size_t ws_size,
                              hipStream_t stream) {
  (void)in_sizes; (void)n_in; (void)out_size; (void)ws_size;
  const float* x    = (const float*)d_in[0];
  const float* Wqa  = (const float*)d_in[1];
  const float* gqa  = (const float*)d_in[2];
  const float* Wqb  = (const float*)d_in[3];
  const float* Wkva = (const float*)d_in[4];
  const float* gkva = (const float*)d_in[5];
  const float* Wkvb = (const float*)d_in[6];
  const float* Wo   = (const float*)d_in[7];
  float* out = (float*)d_out;

  char* ws = (char*)d_ws;
  size_t off = 0;
  auto alloc = [&](size_t bytes) -> char* {
    char* p = ws + off;
    off += (bytes + 255) & ~(size_t)255;
    return p;
  };
  unsigned short* xb    = (unsigned short*)alloc((size_t)BS * Dc * 2);
  unsigned short* WqaT  = (unsigned short*)alloc((size_t)QRc * Dc * 2);
  unsigned short* WqbT  = (unsigned short*)alloc((size_t)QW * QRc * 2);
  unsigned short* WkvaT = (unsigned short*)alloc((size_t)(KVRc + DRc) * Dc * 2);
  unsigned short* WkvbT = (unsigned short*)alloc((size_t)KVW * KVRc * 2);
  unsigned short* WoT   = (unsigned short*)alloc((size_t)Dc * (Hc * DVc) * 2);
  float* qa   = (float*)alloc((size_t)BS * QRc * 4);
  unsigned short* qan  = (unsigned short*)alloc((size_t)BS * QRc * 2);
  float* qbig = (float*)alloc((size_t)BS * QW * 4);
  float* kvc  = (float*)alloc((size_t)BS * (KVRc + DRc) * 4);
  unsigned short* ckvn = (unsigned short*)alloc((size_t)BS * KVRc * 2);
  float* kvbig = (float*)alloc((size_t)BS * KVW * 4);
  unsigned short* qfull = (unsigned short*)alloc((size_t)Bc * Hc * Sc * DFULL * 2);
  unsigned short* kfull = (unsigned short*)alloc((size_t)Bc * Hc * Sc * DFULL * 2);
  unsigned short* vtb   = (unsigned short*)alloc((size_t)Bc * Hc * DVc * Sc * 2);
  unsigned short* ao    = (unsigned short*)alloc((size_t)BS * Hc * DVc * 2);

  // 1) x -> bf16
  {
    unsigned n = (unsigned)BS * Dc;
    convert_f32_to_bf16<<<(n + 255) / 256, 256, 0, stream>>>(x, xb, n);
  }
  // 2) transpose+convert weights to (N,K) bf16
  transpose_f32_to_bf16<<<dim3(QRc / 32, Dc / 32), 256, 0, stream>>>(Wqa, WqaT, Dc, QRc);
  transpose_f32_to_bf16<<<dim3(QW / 32, QRc / 32), 256, 0, stream>>>(Wqb, WqbT, QRc, QW);
  transpose_f32_to_bf16<<<dim3((KVRc + DRc) / 32, Dc / 32), 256, 0, stream>>>(Wkva, WkvaT, Dc, KVRc + DRc);
  transpose_f32_to_bf16<<<dim3(KVW / 32, KVRc / 32), 256, 0, stream>>>(Wkvb, WkvbT, KVRc, KVW);
  transpose_f32_to_bf16<<<dim3(Dc / 32, (Hc * DVc) / 32), 256, 0, stream>>>(Wo, WoT, Hc * DVc, Dc);

  // 3) qa = x @ Wqa  (4096 x 1536, K=2048)
  gemm_bf16_wmma<<<dim3(QRc / 64, BS / 256), 256, 0, stream>>>(xb, WqaT, qa, BS, QRc, Dc);
  // 4) rmsnorm(qa)*gqa -> bf16
  rmsnorm_to_bf16<<<BS, 256, 0, stream>>>(qa, QRc, gqa, qan, QRc);
  // 5) q = qa_n @ Wqb  (4096 x 3072, K=1536)
  gemm_bf16_wmma<<<dim3(QW / 64, BS / 256), 256, 0, stream>>>(qan, WqbT, qbig, BS, QW, QRc);
  // 6) kvc = x @ Wkva  (4096 x 576, K=2048)
  gemm_bf16_wmma<<<dim3((KVRc + DRc) / 64, BS / 256), 256, 0, stream>>>(xb, WkvaT, kvc, BS, KVRc + DRc, Dc);
  // 7) rmsnorm(c_kv)*gkva -> bf16 (first 512 cols of kvc)
  rmsnorm_to_bf16<<<BS, 256, 0, stream>>>(kvc, KVRc + DRc, gkva, ckvn, KVRc);
  // 8) kv = ckv_n @ Wkvb  (4096 x 4096, K=512)
  gemm_bf16_wmma<<<dim3(KVW / 64, BS / 256), 256, 0, stream>>>(ckvn, WkvbT, kvbig, BS, KVW, KVRc);
  // 9) build q_full / k_full (+RoPE) and V^T, bf16
  {
    unsigned n = (unsigned)Bc * Hc * Sc * DFULL;
    build_qfull<<<(n + 255) / 256, 256, 0, stream>>>(qbig, qfull);
    unsigned n2 = (unsigned)Bc * Hc * Sc * (DFULL + DVc);
    build_kfull_v<<<(n2 + 255) / 256, 256, 0, stream>>>(kvbig, kvc, kfull, vtb);
  }
  // 10) causal attention (flash-style, WMMA)
  mla_attention<<<dim3(Sc / 128, Hc, Bc), 256, 0, stream>>>(qfull, kfull, vtb, ao);
  // 11) out = attn_out @ Wo  (4096 x 2048, K=2048) -> fp32 d_out
  gemm_bf16_wmma<<<dim3(Dc / 64, BS / 256), 256, 0, stream>>>(ao, WoT, out, BS, Dc, Hc * DVc);
}
